// SpikeNet_50878182588765
// MI455X (gfx1250) — matrix operand order repaired and checked by reading, not compiled, not run on hardware
//
#include <hip/hip_runtime.h>
#include <math.h>

// ---------------------------------------------------------------------------
// SpikeNet (T=4, N=100000, F=128, K=25, B=10000, SIZES=[5,2], HIDS=[256,128],
// OUT=64, TAU=1 -> LIF state is memoryless: spike = (pre > 1)).
//
// Pipeline per timestep:
//   sample hop0 (threefry)  -> cur1[5B]
//   sample hop1 (threefry)  -> cur2[10B]
//   gather+mean             -> SG0 [60000 x 256] f16   ([self | mean(neigh)])
//   WMMA GEMM + bias + spike-> SPK0 [60000 x 256] f16  (K=256, W0c=[Wl0|Wr0])
//   aggregate spikes        -> SG1 [10000 x 512] f16
//   WMMA GEMM + bias + spike-> Sfin[:, t*128:+128] f16 (K=512)
// Then: WMMA GEMM [10000x512]x[512x64] + bp -> d_out f32.
// ---------------------------------------------------------------------------

#define BN    10000      // batch nodes
#define NN    100000     // graph nodes
#define FF    128        // input features
#define KMAX  25         // padded neighbor table width
#define TT    4

typedef _Float16 v16h __attribute__((ext_vector_type(16)));
typedef float    v8f  __attribute__((ext_vector_type(8)));

union Frag16 { v16h v; uint4 q[2]; };
union Acc8   { v8f  v; float f[8]; };
union Pack4  { _Float16 h[4]; uint2 u; };
union Pack8  { _Float16 h[8]; uint4 q; };

// ------------------------------ threefry2x32 -------------------------------
#define ROTL32(x, r) (((x) << (r)) | ((x) >> (32 - (r))))

__host__ __device__ inline void threefry2x32(unsigned k0, unsigned k1,
                                             unsigned x0, unsigned x1,
                                             unsigned& o0, unsigned& o1) {
  const unsigned ks2 = 0x1BD11BDAu ^ k0 ^ k1;
#define TF_R(r) { x0 += x1; x1 = ROTL32(x1, r); x1 ^= x0; }
  x0 += k0; x1 += k1;
  TF_R(13) TF_R(15) TF_R(26) TF_R(6)  x0 += k1;  x1 += ks2 + 1u;
  TF_R(17) TF_R(29) TF_R(16) TF_R(24) x0 += ks2; x1 += k0  + 2u;
  TF_R(13) TF_R(15) TF_R(26) TF_R(6)  x0 += k0;  x1 += k1  + 3u;
  TF_R(17) TF_R(29) TF_R(16) TF_R(24) x0 += k1;  x1 += ks2 + 4u;
  TF_R(13) TF_R(15) TF_R(26) TF_R(6)  x0 += ks2; x1 += k0  + 5u;
#undef TF_R
  o0 = x0; o1 = x1;
}

__device__ inline float jax_uniform_bits(unsigned k0, unsigned k1, int flat, int n) {
  // JAX: counts = iota(n); pairs (j, j+n/2); out = concat(out0, out1). n is even here.
  const int half = n >> 1;
  unsigned a, b, bits;
  if (flat < half) { threefry2x32(k0, k1, (unsigned)flat, (unsigned)(flat + half), a, b); bits = a; }
  else             { threefry2x32(k0, k1, (unsigned)(flat - half), (unsigned)flat, a, b); bits = b; }
  const unsigned m = (bits >> 9) | 0x3f800000u;       // [1,2)
  float f; __builtin_memcpy(&f, &m, 4);
  return f - 1.0f;                                    // [0,1)
}

// ------------------------- neighbor sampling kernel ------------------------
__global__ void spk_sample_kernel(const int* __restrict__ src, int M, int s1, int s2,
                                  unsigned k1a, unsigned k1b, unsigned k2a, unsigned k2b,
                                  const int* __restrict__ nbr,    const int* __restrict__ deg,
                                  const int* __restrict__ nbr_ev, const int* __restrict__ deg_ev,
                                  int* __restrict__ out) {
  const int stot = s1 + s2;
  const int j = blockIdx.x * blockDim.x + threadIdx.x;
  if (j >= M * stot) return;
  const int i = j / stot, slot = j % stot;
  const int node = src[i];
  float u; int d; const int* tbl;
  if (slot < s1) {
    u = jax_uniform_bits(k1a, k1b, i * s1 + slot, M * s1);
    d = deg[node]; tbl = nbr;
  } else {
    const int s = slot - s1;
    u = jax_uniform_bits(k2a, k2b, i * s2 + s, M * s2);
    d = deg_ev[node]; tbl = nbr_ev;
  }
  int idx = (int)floorf(u * (float)d);
  if (idx > d - 1) idx = d - 1;
  if (idx < 0) idx = 0;
  out[j] = tbl[(size_t)node * KMAX + idx];
}

// ---------------------- gather + hop-mean (layer-0 input) ------------------
// One wave per output row of SG0 [6B x 256] f16: cols 0..127 self, 128..255 mean.
__global__ void spk_gather0_kernel(const float* __restrict__ xt,
                                   const int* __restrict__ nodes,
                                   const int* __restrict__ cur1,
                                   const int* __restrict__ cur2,
                                   _Float16* __restrict__ SG0) {
  const int wave = (int)((blockIdx.x * blockDim.x + threadIdx.x) >> 5);
  const int lane = threadIdx.x & 31;
  if (wave >= 6 * BN) return;
  const int c = lane * 4;
  // self feature row
  const int rs = (wave < BN) ? nodes[wave] : cur1[wave - BN];
  const float4 s = *(const float4*)(xt + (size_t)rs * FF + c);
  Pack4 ps;
  ps.h[0] = (_Float16)s.x; ps.h[1] = (_Float16)s.y;
  ps.h[2] = (_Float16)s.z; ps.h[3] = (_Float16)s.w;
  *(uint2*)(SG0 + (size_t)wave * 256 + c) = ps.u;
  // neighbor mean
  float a0 = 0.f, a1 = 0.f, a2 = 0.f, a3 = 0.f; float inv;
  if (wave < BN) {
    #pragma unroll
    for (int g = 0; g < 5; ++g) {
      const int r = cur1[wave * 5 + g];
      const float4 v = *(const float4*)(xt + (size_t)r * FF + c);
      a0 += v.x; a1 += v.y; a2 += v.z; a3 += v.w;
    }
    inv = 1.0f / 5.0f;
  } else {
    const int jn = wave - BN;
    #pragma unroll
    for (int g = 0; g < 2; ++g) {
      const int r = cur2[jn * 2 + g];
      const float4 v = *(const float4*)(xt + (size_t)r * FF + c);
      a0 += v.x; a1 += v.y; a2 += v.z; a3 += v.w;
    }
    inv = 0.5f;
  }
  Pack4 pn;
  pn.h[0] = (_Float16)(a0 * inv); pn.h[1] = (_Float16)(a1 * inv);
  pn.h[2] = (_Float16)(a2 * inv); pn.h[3] = (_Float16)(a3 * inv);
  *(uint2*)(SG0 + (size_t)wave * 256 + 128 + c) = pn.u;
}

// ----------------- spike aggregation (layer-1 input, [B x 512]) ------------
__global__ void spk_agg1_kernel(const _Float16* __restrict__ SPK0,
                                _Float16* __restrict__ SG1) {
  const int wave = (int)((blockIdx.x * blockDim.x + threadIdx.x) >> 5);
  const int lane = threadIdx.x & 31;
  if (wave >= BN) return;
  const int c = lane * 8;
  // self spikes (copy 16B)
  *(uint4*)(SG1 + (size_t)wave * 512 + c) =
      *(const uint4*)(SPK0 + (size_t)wave * 256 + c);
  // mean over 5 hop rows
  float acc[8] = {0.f, 0.f, 0.f, 0.f, 0.f, 0.f, 0.f, 0.f};
  #pragma unroll
  for (int g = 0; g < 5; ++g) {
    Pack8 ld; ld.q = *(const uint4*)(SPK0 + (size_t)(BN + 5 * wave + g) * 256 + c);
    #pragma unroll
    for (int e = 0; e < 8; ++e) acc[e] += (float)ld.h[e];
  }
  Pack8 st;
  #pragma unroll
  for (int e = 0; e < 8; ++e) st.h[e] = (_Float16)(acc[e] / 5.0f);
  *(uint4*)(SG1 + (size_t)wave * 512 + 256 + c) = st.q;
}

// -------------------------- weight prep (f32 -> f16) -----------------------
__global__ void spk_prep_kernel(const float* __restrict__ Wl0, const float* __restrict__ Wr0,
                                const float* __restrict__ bl0, const float* __restrict__ br0,
                                const float* __restrict__ Wl1, const float* __restrict__ Wr1,
                                const float* __restrict__ bl1, const float* __restrict__ br1,
                                const float* __restrict__ Wp,
                                _Float16* __restrict__ W0c, float* __restrict__ b0c,
                                _Float16* __restrict__ W1c, float* __restrict__ b1c,
                                _Float16* __restrict__ Wpc) {
  const int tid = blockIdx.x * blockDim.x + threadIdx.x;
  const int n0 = 256 * 256, n1 = 128 * 512, n2 = 64 * 512;
  if (tid < n0) {
    const int n = tid >> 8, k = tid & 255;
    W0c[tid] = (_Float16)(k < 128 ? Wl0[n * 128 + k] : Wr0[n * 128 + (k - 128)]);
  } else if (tid < n0 + n1) {
    const int r = tid - n0, n = r >> 9, k = r & 511;
    W1c[r] = (_Float16)(k < 256 ? Wl1[n * 256 + k] : Wr1[n * 256 + (k - 256)]);
  } else if (tid < n0 + n1 + n2) {
    const int r = tid - n0 - n1;
    Wpc[r] = (_Float16)Wp[r];
  } else {
    const int r = tid - (n0 + n1 + n2);
    if (r < 256) b0c[r] = bl0[r] + br0[r];
    else if (r < 256 + 128) b1c[r - 256] = bl1[r - 256] + br1[r - 256];
  }
}

// ------------------------------ WMMA GEMM ----------------------------------
// PRE[M x N] = A[M x K](f16, lda=K) @ W[N x K]^T (f16) + bias(f32)
// outF != null : store f32 PRE (projection head)
// outF == null : store f16 spike = (PRE > 1.0f) into outH (ldo, colOff)
// One wave computes a 16(M) x 64(N) tile; M%16==0, N%64==0, K%32==0.
__global__ void spk_wmma_gemm_kernel(const _Float16* __restrict__ A, int lda,
                                     const _Float16* __restrict__ W,
                                     const float* __restrict__ bias,
                                     int M, int N, int Kdim,
                                     _Float16* __restrict__ outH,
                                     float* __restrict__ outF,
                                     int ldo, int colOff) {
  const int lane = threadIdx.x & 31;
  const int tilesN = N >> 6;
  const int totalTiles = (M >> 4) * tilesN;
  const int tile = blockIdx.x * (blockDim.x >> 5) + (threadIdx.x >> 5);
  if (tile >= totalTiles) return;                     // wave-uniform
  const int tm = tile / tilesN;
  const int tn = tile % tilesN;
  const int rsel = lane & 15;
  const int kselA = (lane >> 4) << 4;                 // A: 16-K per half-wave
  const int kselB = (lane >> 4) << 4;                 // B: 16-K per half-wave

  const _Float16* aRow = A + (size_t)(tm * 16 + rsel) * lda + kselA;
  const _Float16* w0 = W + (size_t)(tn * 64 +  0 + rsel) * Kdim + kselB;
  const _Float16* w1 = W + (size_t)(tn * 64 + 16 + rsel) * Kdim + kselB;
  const _Float16* w2 = W + (size_t)(tn * 64 + 32 + rsel) * Kdim + kselB;
  const _Float16* w3 = W + (size_t)(tn * 64 + 48 + rsel) * Kdim + kselB;

  v8f acc0 = {}; v8f acc1 = {}; v8f acc2 = {}; v8f acc3 = {};
  for (int kk = 0; kk < Kdim; kk += 32) {
    Frag16 fa;
    fa.q[0] = *(const uint4*)(aRow + kk);
    fa.q[1] = *(const uint4*)(aRow + kk + 8);
    Frag16 fb0, fb1, fb2, fb3;
    fb0.q[0] = *(const uint4*)(w0 + kk); fb0.q[1] = *(const uint4*)(w0 + kk + 8);
    fb1.q[0] = *(const uint4*)(w1 + kk); fb1.q[1] = *(const uint4*)(w1 + kk + 8);
    fb2.q[0] = *(const uint4*)(w2 + kk); fb2.q[1] = *(const uint4*)(w2 + kk + 8);
    fb3.q[0] = *(const uint4*)(w3 + kk); fb3.q[1] = *(const uint4*)(w3 + kk + 8);
    acc0 = __builtin_amdgcn_wmma_f32_16x16x32_f16(false, fa.v, false, fb0.v, (short)0, acc0, false, false);
    acc1 = __builtin_amdgcn_wmma_f32_16x16x32_f16(false, fa.v, false, fb1.v, (short)0, acc1, false, false);
    acc2 = __builtin_amdgcn_wmma_f32_16x16x32_f16(false, fa.v, false, fb2.v, (short)0, acc2, false, false);
    acc3 = __builtin_amdgcn_wmma_f32_16x16x32_f16(false, fa.v, false, fb3.v, (short)0, acc3, false, false);
  }

  // D layout: VGPR j -> row = 16*tm + j + 8*(lane>>4), col = 64*tn + 16*c + (lane&15)
  Acc8 u[4]; u[0].v = acc0; u[1].v = acc1; u[2].v = acc2; u[3].v = acc3;
  const int rbase = tm * 16 + ((lane >> 4) << 3);
  #pragma unroll
  for (int c = 0; c < 4; ++c) {
    const int ocol = tn * 64 + c * 16 + rsel;
    const float bv = bias[ocol];
    #pragma unroll
    for (int j = 0; j < 8; ++j) {
      const float v = u[c].f[j] + bv;
      const size_t o = (size_t)(rbase + j) * ldo + colOff + ocol;
      if (outF) outF[o] = v;
      else      outH[o] = (_Float16)((v > 1.0f) ? 1.0f : 0.0f);
    }
  }
}

// ------------------------------- launch ------------------------------------
extern "C" void kernel_launch(void* const* d_in, const int* in_sizes, int n_in,
                              void* d_out, int out_size, void* d_ws, size_t ws_size,
                              hipStream_t stream) {
  (void)in_sizes; (void)n_in; (void)out_size; (void)ws_size;
  const int*   nodes  = (const int*)d_in[0];
  const float* x      = (const float*)d_in[1];
  const int*   nbr    = (const int*)d_in[2];
  const int*   deg    = (const int*)d_in[3];
  const int*   nbr_ev = (const int*)d_in[4];
  const int*   deg_ev = (const int*)d_in[5];
  const float* Wl0 = (const float*)d_in[6];
  const float* Wr0 = (const float*)d_in[7];
  const float* bl0 = (const float*)d_in[8];
  const float* br0 = (const float*)d_in[9];
  const float* Wl1 = (const float*)d_in[10];
  const float* Wr1 = (const float*)d_in[11];
  const float* bl1 = (const float*)d_in[12];
  const float* br1 = (const float*)d_in[13];
  const float* Wp  = (const float*)d_in[14];
  const float* bp  = (const float*)d_in[15];
  float* out = (float*)d_out;

  // workspace carve-up (256B aligned)
  char* base = (char*)d_ws;
  size_t off = 0;
  auto carve = [&](size_t bytes) -> char* {
    char* p = base + off;
    off = (off + bytes + 255) & ~(size_t)255;
    return p;
  };
  _Float16* W0c  = (_Float16*)carve(256 * 256 * 2);
  float*    b0c  = (float*)   carve(256 * 4);
  _Float16* W1c  = (_Float16*)carve(128 * 512 * 2);
  float*    b1c  = (float*)   carve(128 * 4);
  _Float16* Wpc  = (_Float16*)carve(64 * 512 * 2);
  int*      cur1 = (int*)     carve((size_t)5 * BN * 4);
  int*      cur2 = (int*)     carve((size_t)10 * BN * 4);
  _Float16* SG0  = (_Float16*)carve((size_t)6 * BN * 256 * 2);
  _Float16* SPK0 = (_Float16*)carve((size_t)6 * BN * 256 * 2);
  _Float16* SG1  = (_Float16*)carve((size_t)BN * 512 * 2);
  _Float16* Sfin = (_Float16*)carve((size_t)BN * 512 * 2);

  spk_prep_kernel<<<(256 * 256 + 128 * 512 + 64 * 512 + 384 + 255) / 256, 256, 0, stream>>>(
      Wl0, Wr0, bl0, br0, Wl1, Wr1, bl1, br1, Wp, W0c, b0c, W1c, b1c, Wpc);

  for (int t = 0; t < TT; ++t) {
    unsigned k1a, k1b, k2a, k2b;
    // fold_in(key(42), n) == threefry2x32((0,42), (0,n))
    threefry2x32(0u, 42u, 0u, (unsigned)(t * 10 + 0), k1a, k1b);
    threefry2x32(0u, 42u, 0u, (unsigned)(t * 10 + 1), k2a, k2b);
    spk_sample_kernel<<<(BN * 5 + 255) / 256, 256, 0, stream>>>(
        nodes, BN, 2, 3, k1a, k1b, k2a, k2b,
        nbr + (size_t)t * NN * KMAX, deg + (size_t)t * NN,
        nbr_ev + (size_t)t * NN * KMAX, deg_ev + (size_t)t * NN, cur1);

    threefry2x32(0u, 42u, 0u, (unsigned)(t * 10 + 2), k1a, k1b);
    threefry2x32(0u, 42u, 0u, (unsigned)(t * 10 + 3), k2a, k2b);
    spk_sample_kernel<<<(BN * 10 + 255) / 256, 256, 0, stream>>>(
        cur1, BN * 5, 1, 1, k1a, k1b, k2a, k2b,
        nbr + (size_t)t * NN * KMAX, deg + (size_t)t * NN,
        nbr_ev + (size_t)t * NN * KMAX, deg_ev + (size_t)t * NN, cur2);

    spk_gather0_kernel<<<(6 * BN) / 4, 128, 0, stream>>>(
        x + (size_t)t * NN * FF, nodes, cur1, cur2, SG0);

    // layer 0: [60000 x 256] @ [256 x 256] -> spikes SPK0
    spk_wmma_gemm_kernel<<<((6 * BN / 16) * (256 / 64) + 3) / 4, 128, 0, stream>>>(
        SG0, 256, W0c, b0c, 6 * BN, 256, 256, SPK0, nullptr, 256, 0);

    spk_agg1_kernel<<<BN / 4, 128, 0, stream>>>(SPK0, SG1);

    // layer 1: [10000 x 512] @ [512 x 128] -> spikes into Sfin[:, t*128:]
    spk_wmma_gemm_kernel<<<((BN / 16) * (128 / 64) + 3) / 4, 128, 0, stream>>>(
        SG1, 512, W1c, b1c, BN, 128, 512, Sfin, nullptr, 512, t * 128);
  }

  // projection: [10000 x 512] @ [512 x 64] + bp -> f32 out
  spk_wmma_gemm_kernel<<<((BN / 16) * (64 / 64) + 3) / 4, 128, 0, stream>>>(
      Sfin, 512, Wpc, bp, BN, 64, 512, nullptr, out, 64, 0);
}
